// GATEC_79671643341645
// MI455X (gfx1250) — compile-verified, matrix operand order checked by reading, not run on hardware
//
#include <hip/hip_runtime.h>
#include <math.h>

typedef __attribute__((ext_vector_type(16))) _Float16 v16h;
typedef __attribute__((ext_vector_type(8)))  float    v8f;

#define N_NODES   50000
#define N_EDGES   800000
#define K_DIM     128      // all GEMMs have K = 128
#define W_IO      128      // HID*HEADS, inter-layer width
#define HID_D     64
#define OUT_DIM   64
#define NCLS      2
#define NEG_SLOPE 0.2f

// ---------------------------------------------------------------- fill
__global__ void fill_kernel(float* __restrict__ p, float v, int n) {
  int i = blockIdx.x * blockDim.x + threadIdx.x;
  if (i < n) p[i] = v;
}

// ---------------------------------------------------------------- weight pack
// Convert W[128, NCOLS] (f32, row-major) into WMMA B-fragment order (f16):
//   Wp[((kbi*CT + ct)*32 + lane)*16 + elem]
// so each lane's 16-element fragment is one contiguous 32-byte chunk.
// Fragment layout inverse (16-bit B, 32x16 tile): within-block k' in [0,32):
//   half = (k'>>3)&1 ; elem = (k'&7) | ((k'>>4)<<3) ; lane = half*16 + (col&15)
template<int NCOLS>
__global__ void pack_weight_kernel(const float* __restrict__ W, _Float16* __restrict__ Wp) {
  constexpr int CT = NCOLS / 16;
  int idx = blockIdx.x * blockDim.x + threadIdx.x;
  if (idx >= K_DIM * NCOLS) return;
  int k   = idx / NCOLS;
  int col = idx - k * NCOLS;
  int kbi  = k >> 5;
  int kp   = k & 31;
  int half = (kp >> 3) & 1;
  int elem = (kp & 7) | ((kp >> 4) << 3);
  int lane = half * 16 + (col & 15);
  int ct   = col >> 4;
  Wp[(((kbi * CT + ct) * 32) + lane) * 16 + elem] = (_Float16)W[idx];
}

// ---------------------------------------------------------------- WMMA GEMM
// C[M, NCOLS] = A[M, 128] * Wp (+ bias).  Block = 128 threads = 4 waves;
// wave w owns row-tile blockIdx.x*4+w and the FULL output width.
// A fragment: 4x float4 loads + pk converts per K-step, reused across all
// column tiles.  B fragment: single v16h load from the packed buffer.
template<int NCOLS, bool ADD_BIAS>
__global__ __launch_bounds__(128)
void wmma_gemm_kernel(const float* __restrict__ A, const v16h* __restrict__ Wp,
                      const float* __restrict__ bias, float* __restrict__ C, int M) {
  constexpr int CT = NCOLS / 16;
  const int lane  = threadIdx.x & 31;
  const int wave  = threadIdx.x >> 5;
  const int tileM = blockIdx.x * 4 + wave;
  if (tileM * 16 >= M) return;              // wave-uniform; EXEC stays all-ones

  const int half = lane >> 4;               // K-half select per ISA A/B layout
  const int l15  = lane & 15;
  const float* __restrict__ arow = A + (long)(tileM * 16 + l15) * K_DIM;

  v8f acc[CT];
#pragma unroll
  for (int ct = 0; ct < CT; ++ct) acc[ct] = (v8f){};

#pragma unroll
  for (int kbi = 0; kbi < K_DIM / 32; ++kbi) {
    const int kb = kbi * 32;
    // A 16x32 f16 fragment: lanes 0-15 -> K = kb+0..7 & kb+16..23 ;
    // lanes 16-31 -> K = kb+8..15 & kb+24..31.  32B-aligned float4 chunks.
    const float4 p0 = *(const float4*)(arow + kb + half * 8);
    const float4 p1 = *(const float4*)(arow + kb + half * 8 + 4);
    const float4 p2 = *(const float4*)(arow + kb + 16 + half * 8);
    const float4 p3 = *(const float4*)(arow + kb + 16 + half * 8 + 4);
    v16h a;
    a[0]  = (_Float16)p0.x;  a[1]  = (_Float16)p0.y;
    a[2]  = (_Float16)p0.z;  a[3]  = (_Float16)p0.w;
    a[4]  = (_Float16)p1.x;  a[5]  = (_Float16)p1.y;
    a[6]  = (_Float16)p1.z;  a[7]  = (_Float16)p1.w;
    a[8]  = (_Float16)p2.x;  a[9]  = (_Float16)p2.y;
    a[10] = (_Float16)p2.z;  a[11] = (_Float16)p2.w;
    a[12] = (_Float16)p3.x;  a[13] = (_Float16)p3.y;
    a[14] = (_Float16)p3.z;  a[15] = (_Float16)p3.w;

#pragma unroll
    for (int ct = 0; ct < CT; ++ct) {
      const v16h b = Wp[(kbi * CT + ct) * 32 + lane];   // one 32B fragment load
      acc[ct] = __builtin_amdgcn_wmma_f32_16x16x32_f16(false, a, false, b,
                                                       (short)0, acc[ct], false, false);
    }
  }

  // D layout: VGPR r -> row = tileM*16 + half*8 + r, col = ct*16 + (lane&15).
  // M is a multiple of 16 at all call sites (50000 = 3125*16): no row guards.
#pragma unroll
  for (int ct = 0; ct < CT; ++ct) {
    const int col = ct * 16 + l15;
    const float bv = ADD_BIAS ? bias[col] : 0.f;
    float* cptr = C + (long)(tileM * 16 + half * 8) * NCOLS + col;
#pragma unroll
    for (int r = 0; r < 8; ++r) {
      *cptr = acc[ct][r] + bv;
      cptr += NCOLS;
    }
  }
}

// ---------------------------------------------------------------- attention coefficients
__global__ void attn_coef_kernel(const float* __restrict__ feat,
                                 const float* __restrict__ al, const float* __restrict__ ar,
                                 float* __restrict__ el, float* __restrict__ er,
                                 int n, int heads, int dh) {
  int idx = blockIdx.x * blockDim.x + threadIdx.x;
  if (idx >= n * heads) return;
  int node = idx / heads, hd = idx % heads;
  const float* f = feat + (long)node * heads * dh + hd * dh;
  float sl = 0.f, sr = 0.f;
  for (int i = 0; i < dh; ++i) { float v = f[i]; sl += v * al[hd * dh + i]; sr += v * ar[hd * dh + i]; }
  el[idx] = sl; er[idx] = sr;
}

__device__ inline void atomicMaxFloat(float* addr, float val) {
  if (val >= 0.f) atomicMax((int*)addr, __float_as_int(val));
  else            atomicMin((unsigned int*)addr, (unsigned int)__float_as_int(val));
}

// pass 1: e = leaky_relu(el[src]+er[dst]); segment max over dst
__global__ void edge_score_kernel(const int* __restrict__ src, const int* __restrict__ dst,
                                  const float* __restrict__ el, const float* __restrict__ er,
                                  float* __restrict__ ebuf, float* __restrict__ emax,
                                  int e_cnt, int heads) {
  int idx = blockIdx.x * blockDim.x + threadIdx.x;
  if (idx >= e_cnt * heads) return;
  int e = idx / heads, hd = idx % heads;
  int s = src[e], d = dst[e];
  float v = el[s * heads + hd] + er[d * heads + hd];
  v = (v > 0.f) ? v : NEG_SLOPE * v;
  ebuf[idx] = v;
  atomicMaxFloat(&emax[d * heads + hd], v);
}

// pass 2: ex = exp(e - emax[dst]); segment sum over dst
__global__ void edge_exp_kernel(const int* __restrict__ dst, float* __restrict__ ebuf,
                                const float* __restrict__ emax, float* __restrict__ denom,
                                int e_cnt, int heads) {
  int idx = blockIdx.x * blockDim.x + threadIdx.x;
  if (idx >= e_cnt * heads) return;
  int e = idx / heads, hd = idx % heads;
  int d = dst[e];
  float ex = __expf(ebuf[idx] - emax[d * heads + hd]);
  ebuf[idx] = ex;
  atomicAdd(&denom[d * heads + hd], ex);
}

// pass 3: rst[dst] += feat[src] * alpha  (one wave per edge, lane-strided features)
__global__ __launch_bounds__(256)
void edge_aggregate_kernel(const int* __restrict__ src, const int* __restrict__ dst,
                           const float* __restrict__ feat, const float* __restrict__ ebuf,
                           const float* __restrict__ denom, float* __restrict__ rst,
                           int e_cnt, int heads, int dh) {
  int e = blockIdx.x * (blockDim.x >> 5) + (threadIdx.x >> 5);
  int lane = threadIdx.x & 31;
  if (e >= e_cnt) return;
  int s = src[e], d = dst[e];
  int width = heads * dh;
  for (int hd = 0; hd < heads; ++hd) {
    float dn = denom[d * heads + hd];
    float alpha = ebuf[(long)e * heads + hd] / (dn > 0.f ? dn : 1.f);
    for (int i = lane; i < dh; i += 32)
      atomicAdd(&rst[(long)d * width + hd * dh + i],
                feat[(long)s * width + hd * dh + i] * alpha);
  }
}

// finalize: h = elu(rst + b)
__global__ void finalize_kernel(const float* __restrict__ rst, const float* __restrict__ b,
                                float* __restrict__ hout, int n, int width) {
  int idx = blockIdx.x * blockDim.x + threadIdx.x;
  if (idx >= n * width) return;
  int c = idx % width;
  float v = rst[idx] + b[c];
  hout[idx] = (v > 0.f) ? v : (__expf(v) - 1.f);
}

// edge scorer: out[e] = [h[src], h[dst]] @ Ws + bs   (NC = 2, scalar dots)
__global__ void edge_scorer_kernel(const int* __restrict__ src, const int* __restrict__ dst,
                                   const float* __restrict__ h, const float* __restrict__ Ws,
                                   const float* __restrict__ bs, float* __restrict__ out,
                                   int e_cnt) {
  int e = blockIdx.x * blockDim.x + threadIdx.x;
  if (e >= e_cnt) return;
  int s = src[e], d = dst[e];
  const float* hs = h + (long)s * OUT_DIM;
  const float* hd = h + (long)d * OUT_DIM;
  float a0 = bs[0], a1 = bs[1];
#pragma unroll 8
  for (int i = 0; i < OUT_DIM; ++i) {
    a0 += hs[i] * Ws[i * NCLS + 0] + hd[i] * Ws[(OUT_DIM + i) * NCLS + 0];
    a1 += hs[i] * Ws[i * NCLS + 1] + hd[i] * Ws[(OUT_DIM + i) * NCLS + 1];
  }
  out[(long)e * NCLS + 0] = a0;
  out[(long)e * NCLS + 1] = a1;
}

// ---------------------------------------------------------------- launch
extern "C" void kernel_launch(void* const* d_in, const int* in_sizes, int n_in,
                              void* d_out, int out_size, void* d_ws, size_t ws_size,
                              hipStream_t stream) {
  const float* x     = (const float*)d_in[0];
  const int*   src   = (const int*)  d_in[1];
  const int*   dst   = (const int*)  d_in[2];
  const float* W_emb = (const float*)d_in[3];
  const float* b_emb = (const float*)d_in[4];
  const float* Wl[3]  = {(const float*)d_in[5],  (const float*)d_in[9],  (const float*)d_in[13]};
  const float* all[3] = {(const float*)d_in[6],  (const float*)d_in[10], (const float*)d_in[14]};
  const float* arl[3] = {(const float*)d_in[7],  (const float*)d_in[11], (const float*)d_in[15]};
  const float* bl[3]  = {(const float*)d_in[8],  (const float*)d_in[12], (const float*)d_in[16]};
  const float* Ws    = (const float*)d_in[17];
  const float* bs    = (const float*)d_in[18];
  float* out = (float*)d_out;

  const int N = N_NODES, E = N_EDGES;
  float* ws    = (float*)d_ws;
  float* hA    = ws;                        // [N,128] current node features
  float* feat  = hA   + (long)N * W_IO;     // [N,128] transformed features
  float* rst   = feat + (long)N * W_IO;     // [N,128] aggregation accumulator
  float* el    = rst  + (long)N * W_IO;     // [N,2]
  float* er    = el   + (long)N * 2;        // [N,2]
  float* emax  = er   + (long)N * 2;        // [N,2]
  float* denom = emax + (long)N * 2;        // [N,2]
  float* ebuf  = denom+ (long)N * 2;        // [E,2]
  _Float16* Wpack = (_Float16*)(ebuf + (long)E * 2);  // [128*128] packed f16 (32B aligned)

  const int heads_l[3] = {2, 2, 1};
  const int dh_l[3]    = {HID_D, HID_D, OUT_DIM};
  const int gemm_grid  = (N / 16 + 3) / 4;  // 4 row-tiles (waves) per block

  // embedding: hA = x @ W_emb + b_emb
  pack_weight_kernel<128><<<(K_DIM * 128 + 255) / 256, 256, 0, stream>>>(W_emb, Wpack);
  wmma_gemm_kernel<128, true><<<gemm_grid, 128, 0, stream>>>(x, (const v16h*)Wpack, b_emb, hA, N);

  for (int l = 0; l < 3; ++l) {
    const int heads = heads_l[l], dh = dh_l[l], width = heads * dh;

    // feat = hA @ Wl   (no bias on feat; bias applied after aggregation)
    if (width == 128) {
      pack_weight_kernel<128><<<(K_DIM * 128 + 255) / 256, 256, 0, stream>>>(Wl[l], Wpack);
      wmma_gemm_kernel<128, false><<<gemm_grid, 128, 0, stream>>>(hA, (const v16h*)Wpack,
                                                                  nullptr, feat, N);
    } else {
      pack_weight_kernel<64><<<(K_DIM * 64 + 255) / 256, 256, 0, stream>>>(Wl[l], Wpack);
      wmma_gemm_kernel<64,  false><<<gemm_grid, 128, 0, stream>>>(hA, (const v16h*)Wpack,
                                                                  nullptr, feat, N);
    }

    // el/er
    attn_coef_kernel<<<(N * heads + 255) / 256, 256, 0, stream>>>(feat, all[l], arl[l],
                                                                  el, er, N, heads, dh);
    // init segment-max / segment-sum / accumulator
    fill_kernel<<<(N * heads + 255) / 256, 256, 0, stream>>>(emax, -INFINITY, N * heads);
    fill_kernel<<<(N * heads + 255) / 256, 256, 0, stream>>>(denom, 0.f, N * heads);
    fill_kernel<<<(N * width + 255) / 256, 256, 0, stream>>>(rst, 0.f, N * width);
    // softmax over incoming edges
    edge_score_kernel<<<(E * heads + 255) / 256, 256, 0, stream>>>(src, dst, el, er,
                                                                   ebuf, emax, E, heads);
    edge_exp_kernel<<<(E * heads + 255) / 256, 256, 0, stream>>>(dst, ebuf, emax, denom, E, heads);
    // weighted aggregation: one wave per edge
    edge_aggregate_kernel<<<(E + 7) / 8, 256, 0, stream>>>(src, dst, feat, ebuf, denom,
                                                           rst, E, heads, dh);
    // h = elu(rst + b)   (written into hA; last layer leaves [N,64] with ld 64)
    finalize_kernel<<<(N * width + 255) / 256, 256, 0, stream>>>(rst, bl[l], hA, N, width);
  }

  // edge scorer
  edge_scorer_kernel<<<(E + 255) / 256, 256, 0, stream>>>(src, dst, hA, Ws, bs, out, E);
}
